// DPT_82961588289825
// MI455X (gfx1250) — matrix operand order repaired and checked by reading, not compiled, not run on hardware
//
#include <hip/hip_runtime.h>
#include <hip/hip_bf16.h>

typedef __attribute__((ext_vector_type(16))) _Float16 v16h;
typedef __attribute__((ext_vector_type(8)))  float    v8f;

#define DEVFN static __device__ __forceinline__

// ---------------- problem constants ----------------
constexpr int B_ = 4, M_ = 8, L_ = 6, T_ = 6, K_ = 4, E_ = 32, V_ = 2000;
constexpr int L2_ = 2 * L_;        // 12
constexpr int K2_ = 2 * K_;        // 8
constexpr int NS_ = 3;             // NSAMPLE
constexpr int VP_ = 2016;          // vocab padded to multiple of 32 (>= V_)
constexpr int NT_ = V_ / 16;       // 125 full real column tiles (16*125 == V_)
constexpr float EPS_   = 1e-8f;
constexpr float INNER_ = 10.0f * 0.03f;        // INNER_NSTEP * INNER_LR
constexpr float BETASQ2_ = 10.0f;
constexpr float SIGMA_   = 0.31622776601f;     // sqrt(1/BETASQ2)
constexpr float C_INT_   = -0.91893853f;       // 0.5*ln(1/(2*pi))
constexpr float C_NOISE_ =  0.23235401f;       // 0.5*ln(10/(2*pi))

constexpr int NWKE_ = 32 * L2_ * K2_ * L2_;    // 36864 wke rows (bm,i,k,j)
constexpr int NCMB_ = 32 * K2_ * L2_;          // 3072 combos (bm,k,j)

// ---------------- helpers ----------------
DEVFN unsigned long long splitmix(unsigned long long x) {
  x += 0x9E3779B97F4A7C15ull;
  x = (x ^ (x >> 30)) * 0xBF58476D1CE4E5B9ull;
  x = (x ^ (x >> 27)) * 0x94D049BB133111EBull;
  return x ^ (x >> 31);
}
DEVFN float gaussf(unsigned long long key) {
  unsigned long long r = splitmix(key);
  unsigned int u1 = (unsigned int)r, u2 = (unsigned int)(r >> 32);
  float f1 = ((float)u1 + 1.0f) * 2.3283064e-10f;   // (0,1]
  float f2 = (float)u2 * 2.3283064e-10f;
  return __fsqrt_rn(-2.0f * __logf(f1)) * __cosf(6.2831853f * f2);
}
DEVFN unsigned long long nkey(int t, int s, int bm, int i, int k, int j, int e) {
  unsigned long long lin =
      ((((unsigned long long)(bm * L2_ + i)) * K2_ + k) * L2_ + j) * E_ + e;
  return lin + 0x9E3779B97F4A7C15ull * (unsigned long long)(t * NS_ + s + 1);
}

// proposal parent index (priority: i==L+t (t>0) > i==t > i==j > base)
DEVFN int P_of(int i, int j, int t, const int* npar) {
  if (t > 0 && i == L_ + t) return npar[j];
  if (i == t)               return L_ + t;
  if (i == j)               return L_ + t;
  return npar[i];
}
// proposal k index (priority: i==L+t > i==t > i==j > base)
DEVFN int K_of(int i, int k, int j, int t, const int* nark) {
  if (i == L_ + t) return nark[j];
  if (i == t)      return k;
  if (i == j)      return (k + K_) & (K2_ - 1);
  return nark[i];
}
// log graph-proposal prior for sibling j at step t
DEVFN float opc_of(int j, int t) {
  float sum = 0.f, vj = 0.f;
  for (int a = 0; a < L2_; ++a) {
    float mv;
    if (a < L_) { int tm = (t - 1 > 0) ? (t - 1) : 0; mv = (a <= tm) ? 1.f : 0.f; }
    else        { int a2 = a - L_; mv = (a2 <= t - 1 && a2 > 0) ? 1.f : 0.f; }
    float v = mv + EPS_;
    sum += v;
    if (a == j) vj = v;
  }
  return __logf(vj / (sum * (float)K2_));
}

// WMMA fragment index maps (contiguous k = 16*(lane/16)+h, wave32, 16x16x32 f16)
DEVFN int frag_k(int lane, int h) {               // A-matrix K for half h
  return h + ((lane >= 16) ? 16 : 0);
}
DEVFN v8f wmma_f16(v16h a, v16h b, v8f c) {
  return __builtin_amdgcn_wmma_f32_16x16x32_f16(false, a, false, b, (short)0, c,
                                                false, false);
}

// ================= K0: init =================
__global__ void k_init(const float* __restrict__ emb, _Float16* emb16,
                       _Float16* embT16, float* node_ie0, int* npar0, int* nark0,
                       float* lpg0) {
  int idx = blockIdx.x * 256 + threadIdx.x;
  if (idx < VP_ * E_) {
    int v = idx / E_, e = idx % E_;
    float val = (v < V_) ? emb[v * E_ + e] : 0.f;
    emb16[v * E_ + e]   = (_Float16)val;   // (v,e) layout: forward B operand
    embT16[e * VP_ + v] = (_Float16)val;   // (e,v) layout: backward B operand
  }
  if (idx < 32 * L2_ * E_) node_ie0[idx] = 0.f;
  if (idx < 32 * L2_) { npar0[idx] = 0; nark0[idx] = 0; }
  if (idx < 32) lpg0[idx] = 0.f;
}

// ================= K1: wke = gather(node_ie) @ w_k[kidx]; wke[i==T]=0 ======
// one wave per block -> all row indices are wave-uniform (scalar loads)
__global__ void k_wke(const float* __restrict__ node_ie,
                      const int* __restrict__ node_par,
                      const int* __restrict__ node_par_k,
                      const float* __restrict__ w_k, float* wke,
                      _Float16* wke16, int t) {
  int wave = blockIdx.x;
  int lane = threadIdx.x;
  int j = wave % L2_; int r = wave / L2_;
  int k = r % K2_;    r /= K2_;
  int i = r % L2_;    int bm = r / L2_;
  float acc = 0.f;
  if (i != T_) {
    const int* npar = node_par + bm * L2_;
    const int* nark = node_par_k + bm * L2_;
    int p   = P_of(i, j, t, npar);
    int kid = K_of(i, k, j, t, nark);
    const float* v = node_ie + (bm * L2_ + p) * E_;
    const float* W = w_k + kid * E_ * E_;
    #pragma unroll
    for (int e = 0; e < E_; ++e) acc += v[e] * W[e * E_ + lane];
  }
  wke[wave * E_ + lane]   = acc;
  wke16[wave * E_ + lane] = (_Float16)acc;
}

// ================= K2: forward logits (WMMA) + fused online softmax =======
// 1 wave per block, 16 rows; stats accumulated in registers directly from the
// C fragments (no logits buffer): per lane 8 rows x (max, sumexp, tok-logit),
// combined across the 16 lanes of each half with a shfl_xor butterfly.
__global__ void k_fwd(const _Float16* __restrict__ wke16,
                      const _Float16* __restrict__ emb16,
                      const int* __restrict__ tok_ext, float* lp_ext_rows,
                      float* lse_rows, int t) {
  __shared__ int tok_s[16];
  int lane = threadIdx.x;
  int t1 = t + 1;
  int rowBase = blockIdx.x * 16;
  int myrow = rowBase + (lane & 15);             // row = ((bm*K2+k)*L2+j)*t1+i
  int i = myrow % t1; int r = myrow / t1;
  int j = r % L2_; r /= L2_;
  int k = r % K2_; int bm = r / K2_;
  int wrow = ((bm * L2_ + i) * K2_ + k) * L2_ + j;
  if (lane < 16) tok_s[lane] = tok_ext[(bm / M_) * L_ + i];
  __syncthreads();

  v16h a;
  #pragma unroll
  for (int h = 0; h < 16; ++h) a[h] = wke16[wrow * E_ + frag_k(lane, h)];
  int boff = (lane >= 16) ? 16 : 0;
  int mbase = (lane >= 16) ? 8 : 0;
  int col0 = lane & 15;                          // this lane's column phase

  int tokr[8];
  float mx[8], sm[8], tkl[8];
  #pragma unroll
  for (int rr = 0; rr < 8; ++rr) {
    tokr[rr] = tok_s[rr + mbase];
    mx[rr] = -3.4e38f; sm[rr] = 0.f; tkl[rr] = -3.4e38f;
  }

  // double-buffered B tiles over the 125 real column tiles
  v16h bf = *(const v16h*)(emb16 + col0 * E_ + boff);
  for (int c = 0; c < NT_; ++c) {
    v16h bnext = bf;
    if (c + 1 < NT_)
      bnext = *(const v16h*)(emb16 + ((c + 1) * 16 + col0) * E_ + boff);
    if (c + 2 < NT_)
      __builtin_prefetch(emb16 + ((c + 2) * 16 + col0) * E_, 0, 1);
    v8f cz = {};
    v8f cf = wmma_f16(a, bf, cz);
    int col = c * 16 + col0;
    #pragma unroll
    for (int rr = 0; rr < 8; ++rr) {
      float x = cf[rr];
      float mn = fmaxf(mx[rr], x);
      sm[rr] = sm[rr] * __expf(mx[rr] - mn) + __expf(x - mn);
      mx[rr] = mn;
      if (col == tokr[rr]) tkl[rr] = x;
    }
    bf = bnext;
  }

  // butterfly combine across the 16 lanes of each half (masks stay in-half)
  #pragma unroll
  for (int off = 1; off < 16; off <<= 1) {
    #pragma unroll
    for (int rr = 0; rr < 8; ++rr) {
      float om = __shfl_xor(mx[rr], off, 32);
      float os = __shfl_xor(sm[rr], off, 32);
      float ot = __shfl_xor(tkl[rr], off, 32);
      float mn = fmaxf(mx[rr], om);
      sm[rr] = sm[rr] * __expf(mx[rr] - mn) + os * __expf(om - mn);
      mx[rr] = mn;
      tkl[rr] = fmaxf(tkl[rr], ot);
    }
  }
  if ((lane & 15) == 0) {                        // lane 0 -> rows 0-7, lane 16 -> 8-15
    #pragma unroll
    for (int rr = 0; rr < 8; ++rr) {
      int rg = rowBase + rr + mbase;
      float lse = mx[rr] + __logf(sm[rr]);
      lp_ext_rows[rg] = tkl[rr] - lse;
      lse_rows[rg] = lse;
    }
  }
}

// ================= K3: opt_logp (graph prior + lp_ext + internal/noise) ====
__global__ void k_optlogp(const float* __restrict__ lp_ext_rows,
                          const float* __restrict__ wke,
                          const float* __restrict__ node_ie,
                          const float* __restrict__ lp_graph_old,
                          float* opt_logp, float* lp_graph_prop, int t) {
  int combo = blockIdx.x;                         // (bm*K2+k)*L2+j
  int lane = threadIdx.x;
  int t1 = t + 1;
  int j = combo % L2_; int k = (combo / L2_) % K2_; int bm = combo / (K2_ * L2_);
  float lpgp = opc_of(j, t) + lp_graph_old[bm];
  float sext = 0.f;
  for (int i = 0; i <= t; ++i) sext += lp_ext_rows[combo * t1 + i];
  float acc = 0.f;
  int tot = NS_ * t1 * E_;
  for (int idx = lane; idx < tot; idx += 32) {
    int e = idx & 31; int q = idx >> 5;
    int io = q % t1; int s = q / t1;
    int i2 = T_ + io;
    float n = SIGMA_ * gaussf(nkey(t, s, bm, i2, k, j, e));
    int wrow = ((bm * L2_ + i2) * K2_ + k) * L2_ + j;
    float d = node_ie[(bm * L2_ + i2) * E_ + e] + n - wke[wrow * E_ + e];
    acc += C_INT_ - 0.5f * d * d - C_NOISE_ + 0.5f * BETASQ2_ * n * n;
  }
  #pragma unroll
  for (int off = 16; off >= 1; off >>= 1) acc += __shfl_xor(acc, off, 32);
  if (lane == 0) {
    opt_logp[combo] = lpgp + sext + acc * (1.0f / NS_);
    lp_graph_prop[combo] = lpgp;
  }
}

// ================= K4: per-batch top-M (sorted desc, lowest index on ties) =
__global__ void k_topk(const float* __restrict__ opt_logp, float* lp_joint,
                       int* max_idx) {
  int b = blockIdx.x, lane = threadIdx.x;
  const float* v = opt_logp + b * (M_ * K2_ * L2_);
  int chosen[M_];
  for (int sel = 0; sel < M_; ++sel) {
    float bv = -3.4e38f; int bi = 1 << 30;
    for (int idx = lane; idx < M_ * K2_ * L2_; idx += 32) {
      bool used = false;
      for (int s = 0; s < sel; ++s) used |= (chosen[s] == idx);
      if (!used) {
        float val = v[idx];
        if (val > bv || (val == bv && idx < bi)) { bv = val; bi = idx; }
      }
    }
    #pragma unroll
    for (int off = 16; off >= 1; off >>= 1) {
      float ov = __shfl_xor(bv, off, 32);
      int   oi = __shfl_xor(bi, off, 32);
      if (ov > bv || (ov == bv && oi < bi)) { bv = ov; bi = oi; }
    }
    chosen[sel] = bi;
    if (lane == 0) { lp_joint[b * M_ + sel] = bv; max_idx[b * M_ + sel] = bi; }
  }
}

// ================= K5: softmax backward for SELECTED rows only (WMMA) ======
// 1 wave per block, 16 selected rows; recompute logits, p=exp(l-lse),
// dwke = emb[tok] - p @ emb
__global__ void k_bwdext(const int* __restrict__ max_idx,
                         const _Float16* __restrict__ wke16,
                         const _Float16* __restrict__ emb16,
                         const _Float16* __restrict__ embT16,
                         const float* __restrict__ emb,
                         const int* __restrict__ tok_ext,
                         const float* __restrict__ lse_rows, float* dwke_buf,
                         int t) {
  __shared__ float pbuf[16 * 32];
  __shared__ float lse_s[16];
  __shared__ int tok_s[16];
  int lane = threadIdx.x;
  int t1 = t + 1;
  int srow = blockIdx.x * 16 + (lane & 15);      // (b*M+mnew)*t1 + i
  int i = srow % t1; int q = srow / t1;
  int mnew = q & (M_ - 1); int b = q >> 3;
  int n = max_idx[b * M_ + mnew];
  int m = n / (K2_ * L2_); int k = (n / L2_) % K2_; int j = n % L2_;
  int bm = b * M_ + m;
  int orow = ((bm * K2_ + k) * L2_ + j) * t1 + i;
  int wrow = ((bm * L2_ + i) * K2_ + k) * L2_ + j;
  if (lane < 16) { lse_s[lane] = lse_rows[orow]; tok_s[lane] = tok_ext[b * L_ + i]; }
  __syncthreads();

  v16h a;
  #pragma unroll
  for (int h = 0; h < 16; ++h) a[h] = wke16[wrow * E_ + frag_k(lane, h)];
  int boff = (lane >= 16) ? 16 : 0;
  int mbase = (lane >= 16) ? 8 : 0;

  v8f acc0 = {}, acc1 = {};
  for (int kc = 0; kc < VP_ / 32; ++kc) {
    #pragma unroll
    for (int par = 0; par < 2; ++par) {
      int c = kc * 2 + par;
      int N = c * 16 + (lane & 15);
      v16h bf = *(const v16h*)(emb16 + N * E_ + boff);
      v8f cz = {};
      v8f cf = wmma_f16(a, bf, cz);
      int col = c * 16 + (lane & 15);
      #pragma unroll
      for (int rr = 0; rr < 8; ++rr) {
        int Mrow = rr + mbase;
        float p = (col < V_) ? __expf(cf[rr] - lse_s[Mrow]) : 0.f;
        pbuf[Mrow * 32 + par * 16 + (lane & 15)] = p;
      }
    }
    __syncthreads();
    v16h a2;
    #pragma unroll
    for (int h = 0; h < 16; ++h)
      a2[h] = (_Float16)pbuf[(lane & 15) * 32 + frag_k(lane, h)];
    v16h b0 = *(const v16h*)(embT16 + (lane & 15) * VP_ + kc * 32 + boff);
    v16h b1 = *(const v16h*)(embT16 + (16 + (lane & 15)) * VP_ + kc * 32 + boff);
    acc0 = wmma_f16(a2, b0, acc0);
    acc1 = wmma_f16(a2, b1, acc1);
    __syncthreads();
  }
  #pragma unroll
  for (int rr = 0; rr < 8; ++rr) {
    int Mrow = rr + mbase;
    int sg = blockIdx.x * 16 + Mrow;
    int tk = tok_s[Mrow];
    int e0 = lane & 15, e1 = 16 + (lane & 15);
    dwke_buf[sg * E_ + e0] = emb[tk * E_ + e0] - acc0[rr];
    dwke_buf[sg * E_ + e1] = emb[tk * E_ + e1] - acc1[rr];
  }
}

// ================= K6: gradient assembly + inner update + state selection ==
// 1 wave per selected combo (b,mnew); lane = e
__global__ void k_update(const int* __restrict__ max_idx,
                         const float* __restrict__ dwke_buf,
                         const float* __restrict__ wke,
                         const float* __restrict__ w_k,
                         const float* __restrict__ node_ie_old,
                         const int* __restrict__ node_par_old,
                         const int* __restrict__ node_park_old,
                         const float* __restrict__ lp_graph_prop,
                         float* node_ie_new, int* node_par_new,
                         int* node_park_new, float* lp_graph_new, int t) {
  __shared__ float gsh[L2_ * 32];
  __shared__ float dwv[32];
  int combo = blockIdx.x;                         // b*M+mnew
  int lane = threadIdx.x;
  int t1 = t + 1;
  int mnew = combo & (M_ - 1); int b = combo >> 3;
  int n = max_idx[combo];
  int m = n / (K2_ * L2_); int k = (n / L2_) % K2_; int j = n % L2_;
  int bm = b * M_ + m;
  const int* npar = node_par_old + bm * L2_;
  const int* nark = node_park_old + bm * L2_;

  #pragma unroll
  for (int i = 0; i < L2_; ++i) gsh[i * 32 + lane] = 0.f;
  __syncthreads();

  // external chain: i' in [0,t], dnie = dwke_ext . W[kid]^T, scatter at P(i')
  for (int i2 = 0; i2 <= t; ++i2) {
    int s = combo * t1 + i2;
    int kid = K_of(i2, k, j, t, nark);
    const float* Wr = w_k + kid * E_ * E_ + lane * E_;
    float acc = 0.f;
    #pragma unroll
    for (int f = 0; f < E_; ++f) acc += dwke_buf[s * E_ + f] * Wr[f];
    int p = P_of(i2, j, t, npar);
    gsh[p * 32 + lane] += acc;
  }
  // internal rows: direct term for i' in [T, T+t]; chain only for i' > T
  for (int i2 = T_; i2 <= T_ + t; ++i2) {
    float nb = 0.f;
    for (int ss = 0; ss < NS_; ++ss) nb += gaussf(nkey(t, ss, bm, i2, k, j, lane));
    nb *= SIGMA_ * (1.0f / NS_);                  // mean noise
    int wrow = ((bm * L2_ + i2) * K2_ + k) * L2_ + j;
    float diff = node_ie_old[(bm * L2_ + i2) * E_ + lane] - wke[wrow * E_ + lane] + nb;
    gsh[i2 * 32 + lane] += -diff;                 // direct d/d(nie)
    if (i2 > T_) {                                // chain (wke[T] grad blocked)
      dwv[lane] = diff;
      __syncthreads();
      int kid = K_of(i2, k, j, t, nark);
      const float* Wr = w_k + kid * E_ * E_ + lane * E_;
      float acc = 0.f;
      #pragma unroll
      for (int f = 0; f < E_; ++f) acc += dwv[f] * Wr[f];
      int p = P_of(i2, j, t, npar);
      gsh[p * 32 + lane] += acc;
      __syncthreads();
    }
  }
  __syncthreads();

  // apply inner update and select new state
  int bmn = b * M_ + mnew;
  for (int i = 0; i < L2_; ++i)
    node_ie_new[(bmn * L2_ + i) * E_ + lane] =
        node_ie_old[(bm * L2_ + i) * E_ + lane] + INNER_ * gsh[i * 32 + lane];
  if (lane < L2_) {
    node_par_new[bmn * L2_ + lane]  = P_of(lane, j, t, npar);
    node_park_new[bmn * L2_ + lane] = K_of(lane, k, j, t, nark);
  }
  if (lane == 0)
    lp_graph_new[combo] = lp_graph_prop[(bm * K2_ + k) * L2_ + j];
}

// ================= K7: pack outputs =================
__global__ void k_out(const float* __restrict__ lp_joint,
                      const float* __restrict__ lp_graph,
                      const float* __restrict__ node_ie,
                      const int* __restrict__ node_par,
                      const int* __restrict__ node_park, float* out) {
  int idx = blockIdx.x * 256 + threadIdx.x;
  const int NIE = 32 * L2_ * E_;                  // 12288
  const int NP  = 32 * L2_;                       // 384
  if (idx < 32)                        out[idx] = lp_joint[idx];
  else if (idx < 64)                   out[idx] = lp_graph[idx - 32];
  else if (idx < 64 + NIE)             out[idx] = node_ie[idx - 64];
  else if (idx < 64 + NIE + NP)        out[idx] = (float)node_par[idx - 64 - NIE];
  else if (idx < 64 + NIE + 2 * NP)    out[idx] = (float)node_park[idx - 64 - NIE - NP];
}

// ================= host =================
extern "C" void kernel_launch(void* const* d_in, const int* in_sizes, int n_in,
                              void* d_out, int out_size, void* d_ws, size_t ws_size,
                              hipStream_t stream) {
  (void)in_sizes; (void)n_in; (void)out_size; (void)ws_size;
  const int*   tok = (const int*)d_in[0];
  // d_in[1] = enc_inputs_len (unused, as in reference)
  const float* emb = (const float*)d_in[2];
  const float* wk  = (const float*)d_in[3];

  char* ws = (char*)d_ws;
  auto alloc = [&](size_t bytes) -> void* {
    void* p = (void*)ws;
    ws += (bytes + 255) & ~(size_t)255;
    return p;
  };
  _Float16* emb16  = (_Float16*)alloc((size_t)VP_ * E_ * 2);
  _Float16* embT16 = (_Float16*)alloc((size_t)E_ * VP_ * 2);
  float*    wke    = (float*)alloc((size_t)NWKE_ * E_ * 4);
  _Float16* wke16  = (_Float16*)alloc((size_t)NWKE_ * E_ * 2);
  float* lp_ext_rows = (float*)alloc((size_t)NCMB_ * 6 * 4);
  float* lse_rows    = (float*)alloc((size_t)NCMB_ * 6 * 4);
  float* opt_logp    = (float*)alloc((size_t)NCMB_ * 4);
  float* lp_graph_prop = (float*)alloc((size_t)NCMB_ * 4);
  float* dwke_buf    = (float*)alloc((size_t)32 * 6 * E_ * 4);
  float* lp_graph_s[2] = {(float*)alloc(32 * 4), (float*)alloc(32 * 4)};
  float* node_ie_s[2]  = {(float*)alloc(32 * L2_ * E_ * 4),
                          (float*)alloc(32 * L2_ * E_ * 4)};
  int* node_par_s[2]   = {(int*)alloc(32 * L2_ * 4), (int*)alloc(32 * L2_ * 4)};
  int* node_park_s[2]  = {(int*)alloc(32 * L2_ * 4), (int*)alloc(32 * L2_ * 4)};
  float* lp_joint = (float*)alloc(32 * 4);
  int*   max_idx  = (int*)alloc(32 * 4);

  k_init<<<(VP_ * E_ + 255) / 256, 256, 0, stream>>>(
      emb, emb16, embT16, node_ie_s[0], node_par_s[0], node_park_s[0],
      lp_graph_s[0]);

  int cur = 0;
  for (int t = 0; t < 6; ++t) {
    int t1 = t + 1;
    k_wke<<<NWKE_, 32, 0, stream>>>(node_ie_s[cur], node_par_s[cur],
                                    node_park_s[cur], wk, wke, wke16, t);
    k_fwd<<<(NCMB_ / 16) * t1, 32, 0, stream>>>(wke16, emb16, tok, lp_ext_rows,
                                                lse_rows, t);
    k_optlogp<<<NCMB_, 32, 0, stream>>>(lp_ext_rows, wke, node_ie_s[cur],
                                        lp_graph_s[cur], opt_logp,
                                        lp_graph_prop, t);
    k_topk<<<B_, 32, 0, stream>>>(opt_logp, lp_joint, max_idx);
    k_bwdext<<<(32 * t1) / 16, 32, 0, stream>>>(max_idx, wke16, emb16, embT16,
                                                emb, tok, lse_rows, dwke_buf, t);
    k_update<<<32, 32, 0, stream>>>(
        max_idx, dwke_buf, wke, wk, node_ie_s[cur], node_par_s[cur],
        node_park_s[cur], lp_graph_prop, node_ie_s[1 - cur],
        node_par_s[1 - cur], node_park_s[1 - cur], lp_graph_s[1 - cur], t);
    cur ^= 1;
  }
  k_out<<<(13120 + 255) / 256, 256, 0, stream>>>(
      lp_joint, lp_graph_s[cur], node_ie_s[cur], node_par_s[cur],
      node_park_s[cur], (float*)d_out);
}
